// TimeSeriesTransformer_43404939493876
// MI455X (gfx1250) — matrix operand order repaired and checked by reading, not compiled, not run on hardware
//
#include <hip/hip_runtime.h>
#include <math.h>

// ---------------------------------------------------------------------------
// TimeSeriesTransformer forward for MI455X (gfx1250, wave32, WMMA).
// bf16 operands on the matrix path (f32 residual stream), weights pre-packed
// into WMMA B-fragment order so every fragment is a b128 load. GEMM uses a
// 2x2 register tile per wave (4 WMMAs per 8 b128 loads). Attention is fused
// flash-style with transposed V for contiguous P@V fragments. ~116 MB ws.
// ---------------------------------------------------------------------------

typedef __attribute__((ext_vector_type(16))) __bf16 v16bf;
typedef __attribute__((ext_vector_type(8)))  __bf16 v8bf;
typedef __attribute__((ext_vector_type(8)))  float  v8f;

__device__ __forceinline__ v16bf cat16(v8bf lo, v8bf hi) {
    return __builtin_shufflevector(lo, hi, 0, 1, 2, 3, 4, 5, 6, 7,
                                           8, 9, 10, 11, 12, 13, 14, 15);
}

__device__ __forceinline__ float gelu_exact(float x) {
    return 0.5f * x * (1.0f + erff(x * 0.70710678118654752f));
}

// ---------------------------------------------------------------------------
// Weight pack: W[K,N] f32 (row-major) -> bf16 WMMA B-fragment order.
// Flat layout: (((kt*(N/16) + nt)*32 + lane)*16 + e)
//   element = W[kt*32 + (lane>>4)*8 + (e&7) + (e>>3)*16][nt*16 + (lane&15)]
// ---------------------------------------------------------------------------
__global__ void pack_w(const float* __restrict__ W, __bf16* __restrict__ Wp,
                       int K, int N)
{
    int idx = blockIdx.x * blockDim.x + threadIdx.x;
    if (idx >= K * N) return;
    int e    = idx & 15;
    int lane = (idx >> 4) & 31;
    int rest = idx >> 9;
    int ntiles = N >> 4;
    int nt = rest % ntiles;
    int kt = rest / ntiles;
    int k = kt * 32 + ((lane >> 4) << 3) + (e & 7) + ((e >> 3) << 4);
    int n = nt * 16 + (lane & 15);
    Wp[idx] = (__bf16)W[(size_t)k * N + n];
}

// f32 -> bf16 elementwise convert (for the raw input x)
__global__ void cvt_bf16(const float* __restrict__ in, __bf16* __restrict__ out, int n)
{
    int i = blockIdx.x * blockDim.x + threadIdx.x;
    if (i < n) out[i] = (__bf16)in[i];
}

// ---------------------------------------------------------------------------
// GEMM: C[M,N] = act( A[M,K]bf16 @ Wp(packed)bf16 + bias + resid(f32) )
// Wave computes a 32x32 tile (2x2 subtiles): per K-step 8 b128 loads feed
// 4 independent WMMAs. Block = 4 waves (2x2) = 64x64 tile.
// Output: Cf (f32) and/or Cb (bf16); vtrans=1 writes Cb as V^T[(b,h,d),s].
// ---------------------------------------------------------------------------
__global__ __launch_bounds__(128) void gemm_bf16pk(
    const __bf16* __restrict__ A, const __bf16* __restrict__ Wp,
    const float* __restrict__ bias, const float* __restrict__ resid,
    float* __restrict__ Cf, __bf16* __restrict__ Cb,
    int M, int K, int N, int act, int vtrans, int S, int H)
{
    const int lane   = threadIdx.x & 31;
    const int wave   = threadIdx.x >> 5;
    const int laneHi = lane >> 4;
    const int lanelo = lane & 15;
    const int n0 = blockIdx.x * 64 + (wave & 1) * 32;   // wave-uniform
    const int m0 = blockIdx.y * 64 + (wave >> 1) * 32;
    if (n0 >= N || m0 >= M) return;                     // uniform: EXEC full

    const __bf16* ap0  = A + (size_t)(m0 + lanelo) * K + laneHi * 8;
    const __bf16* ap1  = ap0 + (size_t)16 * K;
    const __bf16* wpp0 = Wp + ((size_t)(n0 >> 4) * 32 + lane) * 16;
    const __bf16* wpp1 = wpp0 + 512;                    // next 16-col tile
    const size_t wstride = (size_t)(N >> 4) * 512;      // elements per K-step

    v8f acc00 = {}, acc01 = {}, acc10 = {}, acc11 = {};
    for (int kt = 0; kt < (K >> 5); ++kt) {
        __builtin_prefetch((const void*)(wpp0 + wstride), 0, 3);
        __builtin_prefetch((const void*)(wpp1 + wstride), 0, 3);
        __builtin_prefetch((const void*)(ap0 + 32), 0, 3);
        __builtin_prefetch((const void*)(ap1 + 32), 0, 3);
        v16bf a0 = cat16(*(const v8bf*)(ap0), *(const v8bf*)(ap0 + 16));
        v16bf a1 = cat16(*(const v8bf*)(ap1), *(const v8bf*)(ap1 + 16));
        v16bf b0 = cat16(*(const v8bf*)(wpp0), *(const v8bf*)(wpp0 + 8));
        v16bf b1 = cat16(*(const v8bf*)(wpp1), *(const v8bf*)(wpp1 + 8));
        acc00 = __builtin_amdgcn_wmma_f32_16x16x32_bf16(false, a0, false, b0, (short)0, acc00, false, false);
        acc01 = __builtin_amdgcn_wmma_f32_16x16x32_bf16(false, a0, false, b1, (short)0, acc01, false, false);
        acc10 = __builtin_amdgcn_wmma_f32_16x16x32_bf16(false, a1, false, b0, (short)0, acc10, false, false);
        acc11 = __builtin_amdgcn_wmma_f32_16x16x32_bf16(false, a1, false, b1, (short)0, acc11, false, false);
        ap0  += 32; ap1  += 32;
        wpp0 += wstride; wpp1 += wstride;
    }

    v8f accs[2][2] = { { acc00, acc01 }, { acc10, acc11 } };
#pragma unroll
    for (int mi = 0; mi < 2; ++mi)
#pragma unroll
    for (int ni = 0; ni < 2; ++ni)
#pragma unroll
    for (int v = 0; v < 8; ++v) {                       // C: M = v + 8*laneHi
        int row = m0 + mi * 16 + v + 8 * laneHi;
        int col = n0 + ni * 16 + lanelo;
        float x = accs[mi][ni][v];
        if (bias)  x += bias[col];
        if (resid) x += resid[(size_t)row * N + col];
        if (act == 1) x = gelu_exact(x);
        if (Cf) Cf[(size_t)row * N + col] = x;
        if (Cb) {
            if (!vtrans) {
                Cb[(size_t)row * N + col] = (__bf16)x;
            } else {                                    // V^T: [(b*H+h)*64+dk][s]
                int bi = row / S, s = row % S;
                int hh = col >> 6, dk = col & 63;
                Cb[(((size_t)bi * H + hh) * 64 + dk) * S + s] = (__bf16)x;
            }
        }
    }
}

// ---------------------------------------------------------------------------
// Fused causal flash attention, all-bf16 fragments, contiguous b128 loads.
// Q,K: (B*S, H*64) bf16 row-major.  Vt: (B*H*64, S) bf16.  Out: bf16 (B*S,H*64).
// One wave per (b, h, 16-query tile), streaming softmax over 32-key blocks.
// ---------------------------------------------------------------------------
__global__ __launch_bounds__(32) void flash_attn_wmma(
    const __bf16* __restrict__ Q, const __bf16* __restrict__ Kb,
    const __bf16* __restrict__ Vt, __bf16* __restrict__ Out, int S, int H)
{
    const int bh = blockIdx.x;
    const int b  = bh / H, h = bh % H;
    const int qt = blockIdx.y;
    const int lane   = threadIdx.x & 31;
    const int laneHi = lane >> 4;
    const int lanelo = lane & 15;
    const int D = H * 64;
    const size_t base  = (size_t)b * S * D + (size_t)h * 64;  // Q/K/Out row base
    const size_t vbase = ((size_t)bh * 64) * S;               // V^T head base

    // Q tile (16x64) as two A-fragments: two b128 loads each.
    const int qrow = qt * 16 + lanelo;
    const __bf16* qp = Q + base + (size_t)qrow * D + laneHi * 16;
    v16bf qa0 = cat16(*(const v8bf*)(qp),      *(const v8bf*)(qp + 8));
    v16bf qa1 = cat16(*(const v8bf*)(qp + 32), *(const v8bf*)(qp + 40));

    float m_i[8], l_i[8];
#pragma unroll
    for (int v = 0; v < 8; ++v) { m_i[v] = -1e30f; l_i[v] = 0.0f; }
    v8f o[4] = {};

    __shared__ __bf16 pb[16][32];                  // P tile staging (bf16)

    const int nkb = (qt * 16 + 16 + 31) / 32;
    for (int kb = 0; kb < nkb; ++kb) {
        const int kbase = kb * 32;

        // scores: S = Q @ K^T. K row-major(key,d) == column-major B -> contiguous.
        v8f s0 = {}, s1 = {};
#pragma unroll
        for (int dc = 0; dc < 2; ++dc) {
            const __bf16* kp0 = Kb + base + (size_t)(kbase + lanelo) * D
                                + dc * 32 + laneHi * 16;
            const __bf16* kp1 = kp0 + (size_t)16 * D;
            v16bf bk0 = cat16(*(const v8bf*)(kp0), *(const v8bf*)(kp0 + 8));
            v16bf bk1 = cat16(*(const v8bf*)(kp1), *(const v8bf*)(kp1 + 8));
            v16bf qa = dc ? qa1 : qa0;
            s0 = __builtin_amdgcn_wmma_f32_16x16x32_bf16(false, qa, false, bk0, (short)0, s0, false, false);
            s1 = __builtin_amdgcn_wmma_f32_16x16x32_bf16(false, qa, false, bk1, (short)0, s1, false, false);
        }

        // streaming softmax (rows live in (v, lane-half); reduce across 16 lanes)
        const float sc = 0.125f;                   // 1/sqrt(64)
#pragma unroll
        for (int v = 0; v < 8; ++v) {
            int row  = qt * 16 + v + 8 * laneHi;
            int key0 = kbase + lanelo, key1 = kbase + 16 + lanelo;
            float a0 = (key0 <= row) ? s0[v] * sc : -1e30f;
            float a1 = (key1 <= row) ? s1[v] * sc : -1e30f;
            float mx = fmaxf(a0, a1);
#pragma unroll
            for (int msk = 8; msk >= 1; msk >>= 1) mx = fmaxf(mx, __shfl_xor(mx, msk));
            float mnew = fmaxf(m_i[v], mx);
            float corr = __expf(m_i[v] - mnew);
            float p0 = __expf(a0 - mnew), p1 = __expf(a1 - mnew);
            float rs = p0 + p1;
#pragma unroll
            for (int msk = 8; msk >= 1; msk >>= 1) rs += __shfl_xor(rs, msk);
            l_i[v] = l_i[v] * corr + rs;
            m_i[v] = mnew;
            o[0][v] *= corr; o[1][v] *= corr; o[2][v] *= corr; o[3][v] *= corr;
            int r = v + 8 * laneHi;
            pb[r][lanelo]      = (__bf16)p0;
            pb[r][16 + lanelo] = (__bf16)p1;
        }
        __syncthreads();

        // P (16x32) as A-fragment: two b128 LDS reads per lane.
        v16bf pa = cat16(*(const v8bf*)&pb[lanelo][laneHi * 16],
                         *(const v8bf*)&pb[lanelo][laneHi * 16 + 8]);

        // O += P @ V. V^T rows (fixed d) are contiguous over keys -> b128 loads.
#pragma unroll
        for (int nt = 0; nt < 4; ++nt) {
            const __bf16* vp = Vt + vbase + (size_t)(nt * 16 + lanelo) * S
                               + kbase + laneHi * 16;
            v16bf bv = cat16(*(const v8bf*)(vp), *(const v8bf*)(vp + 8));
            o[nt] = __builtin_amdgcn_wmma_f32_16x16x32_bf16(
                false, pa, false, bv, (short)0, o[nt], false, false);
        }
        __syncthreads();
    }

#pragma unroll
    for (int nt = 0; nt < 4; ++nt)
#pragma unroll
        for (int v = 0; v < 8; ++v) {
            int row = qt * 16 + v + 8 * laneHi;
            int d   = nt * 16 + lanelo;
            Out[base + (size_t)row * D + d] = (__bf16)(o[nt][v] / l_i[v]);
        }
}

// ---------------------------------------------------------------------------
// Row LayerNorm. Input f32; writes f32 (outF) and/or bf16 (outB).
// addPE adds sinusoidal positional encoding after the affine transform.
// ---------------------------------------------------------------------------
__global__ __launch_bounds__(256) void ln_kernel(
    const float* __restrict__ in, const float* __restrict__ w,
    const float* __restrict__ b, float* __restrict__ outF,
    __bf16* __restrict__ outB, int D, int addPE, int S)
{
    __shared__ float ssum[9], ssq[9];
    const int row = blockIdx.x;
    const float* xr = in + (size_t)row * D;
    float s = 0.0f, s2 = 0.0f;
    for (int i = threadIdx.x; i < D; i += blockDim.x) {
        float t = xr[i]; s += t; s2 += t * t;
    }
#pragma unroll
    for (int msk = 16; msk >= 1; msk >>= 1) {
        s  += __shfl_xor(s,  msk);
        s2 += __shfl_xor(s2, msk);
    }
    if ((threadIdx.x & 31) == 0) { ssum[threadIdx.x >> 5] = s; ssq[threadIdx.x >> 5] = s2; }
    __syncthreads();
    if (threadIdx.x == 0) {
        float a = 0.0f, c = 0.0f;
        for (int i = 0; i < 8; ++i) { a += ssum[i]; c += ssq[i]; }
        ssum[8] = a; ssq[8] = c;
    }
    __syncthreads();
    const float mean = ssum[8] / (float)D;
    const float var  = ssq[8] / (float)D - mean * mean;
    const float inv  = rsqrtf(var + 1e-5f);
    const float pos  = (float)(row % S);
    for (int i = threadIdx.x; i < D; i += blockDim.x) {
        float t = (xr[i] - mean) * inv * w[i] + b[i];
        if (addPE) {
            float divv = __expf(-(float)(i & ~1) * (9.210340371976184f / (float)D));
            float ang  = pos * divv;
            t += (i & 1) ? __cosf(ang) : __sinf(ang);
        }
        if (outF) outF[(size_t)row * D + i] = t;
        if (outB) outB[(size_t)row * D + i] = (__bf16)t;
    }
}

// Gather last time step: pooled[b, d] = in[b*S + S-1, d]
__global__ void pool_last(const float* __restrict__ in, float* __restrict__ out,
                          int B, int S, int D)
{
    int idx = blockIdx.x * blockDim.x + threadIdx.x;
    if (idx >= B * D) return;
    int b = idx / D, d = idx % D;
    out[idx] = in[((size_t)b * S + (S - 1)) * D + d];
}

// Tiny head GEMM: out[m,n] = act(in[m,:K] @ W[:,n] + bias[n]); act 0/1(gelu)/2(sigmoid)
__global__ void dense_small(const float* __restrict__ in, const float* __restrict__ W,
                            const float* __restrict__ bias, float* __restrict__ out,
                            int M, int K, int N, int act)
{
    int idx = blockIdx.x * blockDim.x + threadIdx.x;
    if (idx >= M * N) return;
    int m = idx / N, n = idx % N;
    float s = bias ? bias[n] : 0.0f;
    for (int k = 0; k < K; ++k) s += in[(size_t)m * K + k] * W[(size_t)k * N + n];
    if (act == 1)      s = gelu_exact(s);
    else if (act == 2) s = 1.0f / (1.0f + __expf(-s));
    out[idx] = s;
}

// ---------------------------------------------------------------------------
extern "C" void kernel_launch(void* const* d_in, const int* in_sizes, int n_in,
                              void* d_out, int out_size, void* d_ws, size_t ws_size,
                              hipStream_t stream)
{
    (void)in_sizes; (void)n_in; (void)out_size; (void)ws_size;
    const int B = 16, S = 1024, NF = 64, D = 512, H = 8, L = 6, DFF = 2048;
    const int MS = B * S;                          // 16384 rows

    const float* x       = (const float*)d_in[0];
    const float* in_w    = (const float*)d_in[1];
    const float* in_b    = (const float*)d_in[2];
    const float* in_ln_w = (const float*)d_in[3];
    const float* in_ln_b = (const float*)d_in[4];
    const float* Wq      = (const float*)d_in[5];
    const float* Wk      = (const float*)d_in[6];
    const float* Wv      = (const float*)d_in[7];
    const float* Wo      = (const float*)d_in[8];
    const float* ln1_w   = (const float*)d_in[9];
    const float* ln1_b   = (const float*)d_in[10];
    const float* ln2_w   = (const float*)d_in[11];
    const float* ln2_b   = (const float*)d_in[12];
    const float* ff_w1   = (const float*)d_in[13];
    const float* ff_b1   = (const float*)d_in[14];
    const float* ff_w2   = (const float*)d_in[15];
    const float* ff_b2   = (const float*)d_in[16];
    const float* fin_ln_w= (const float*)d_in[17];
    const float* fin_ln_b= (const float*)d_in[18];
    const float* cls_w1  = (const float*)d_in[19];
    const float* cls_b1  = (const float*)d_in[20];
    const float* cls_w2  = (const float*)d_in[21];
    const float* cls_b2  = (const float*)d_in[22];
    const float* reg_w1  = (const float*)d_in[23];
    const float* reg_b1  = (const float*)d_in[24];
    const float* reg_w2  = (const float*)d_in[25];
    const float* reg_b2  = (const float*)d_in[26];
    const float* conf_w1 = (const float*)d_in[27];
    const float* conf_b1 = (const float*)d_in[28];
    const float* conf_w2 = (const float*)d_in[29];
    const float* conf_b2 = (const float*)d_in[30];

    // Workspace layout (bytes).
    char* p = (char*)d_ws;
    float*  h   = (float*)p;            p += (size_t)MS * D * 4;      // 32 MB
    __bf16* hnb = (__bf16*)p;           p += (size_t)MS * D * 2;      // 16 MB
    char*   big = p;                    p += (size_t)MS * DFF * 2;    // 64 MB union
    __bf16* qb  = (__bf16*)big;
    __bf16* kbf = qb  + (size_t)MS * D;
    __bf16* vT  = kbf + (size_t)MS * D;
    __bf16* aob = vT  + (size_t)MS * D;
    __bf16* ff1b = (__bf16*)big;                    // after attention phase
    float*  bigf = (float*)big;                     // f32 scratch (pre-LN / final)
    __bf16* xb  = (__bf16*)p;           p += (size_t)MS * NF * 2;     // 2 MB
    __bf16* wp  = (__bf16*)p;           p += (size_t)D * DFF * 2;     // 2 MB pack buf
    float* pooled = (float*)p;          p += (size_t)B * D * 4;
    float* hid    = (float*)p;

    const dim3 blkG(128);
    const dim3 grdD(D / 64, MS / 64);              // 64x64 block tiles
    const dim3 grdF(DFF / 64, MS / 64);
    const int PKT = 256;

    // ---- input projection + LN + positional encoding ----
    cvt_bf16<<<(MS * NF + PKT - 1) / PKT, PKT, 0, stream>>>(x, xb, MS * NF);
    pack_w<<<(NF * D + PKT - 1) / PKT, PKT, 0, stream>>>(in_w, wp, NF, D);
    gemm_bf16pk<<<grdD, blkG, 0, stream>>>(xb, wp, in_b, nullptr, bigf, nullptr,
                                           MS, NF, D, 0, 0, S, H);
    ln_kernel<<<MS, 256, 0, stream>>>(bigf, in_ln_w, in_ln_b, h, nullptr, D, 1, S);

    for (int l = 0; l < L; ++l) {
        const float* wq = Wq + (size_t)l * D * D;
        const float* wk = Wk + (size_t)l * D * D;
        const float* wv = Wv + (size_t)l * D * D;
        const float* wo = Wo + (size_t)l * D * D;

        ln_kernel<<<MS, 256, 0, stream>>>(h, ln1_w + (size_t)l * D, ln1_b + (size_t)l * D,
                                          nullptr, hnb, D, 0, S);

        pack_w<<<(D * D + PKT - 1) / PKT, PKT, 0, stream>>>(wq, wp, D, D);
        gemm_bf16pk<<<grdD, blkG, 0, stream>>>(hnb, wp, nullptr, nullptr, nullptr, qb,
                                               MS, D, D, 0, 0, S, H);
        pack_w<<<(D * D + PKT - 1) / PKT, PKT, 0, stream>>>(wk, wp, D, D);
        gemm_bf16pk<<<grdD, blkG, 0, stream>>>(hnb, wp, nullptr, nullptr, nullptr, kbf,
                                               MS, D, D, 0, 0, S, H);
        pack_w<<<(D * D + PKT - 1) / PKT, PKT, 0, stream>>>(wv, wp, D, D);
        gemm_bf16pk<<<grdD, blkG, 0, stream>>>(hnb, wp, nullptr, nullptr, nullptr, vT,
                                               MS, D, D, 0, 1, S, H);   // transposed V

        flash_attn_wmma<<<dim3(B * H, S / 16), 32, 0, stream>>>(qb, kbf, vT, aob, S, H);

        pack_w<<<(D * D + PKT - 1) / PKT, PKT, 0, stream>>>(wo, wp, D, D);
        gemm_bf16pk<<<grdD, blkG, 0, stream>>>(aob, wp, nullptr, h, h, nullptr,
                                               MS, D, D, 0, 0, S, H);   // h += ao @ Wo

        ln_kernel<<<MS, 256, 0, stream>>>(h, ln2_w + (size_t)l * D, ln2_b + (size_t)l * D,
                                          nullptr, hnb, D, 0, S);
        pack_w<<<(D * DFF + PKT - 1) / PKT, PKT, 0, stream>>>(ff_w1 + (size_t)l * D * DFF, wp, D, DFF);
        gemm_bf16pk<<<grdF, blkG, 0, stream>>>(hnb, wp, ff_b1 + (size_t)l * DFF, nullptr,
                                               nullptr, ff1b, MS, D, DFF, 1, 0, S, H);
        pack_w<<<(DFF * D + PKT - 1) / PKT, PKT, 0, stream>>>(ff_w2 + (size_t)l * DFF * D, wp, DFF, D);
        gemm_bf16pk<<<grdD, blkG, 0, stream>>>(ff1b, wp, ff_b2 + (size_t)l * D, h, h, nullptr,
                                               MS, DFF, D, 0, 0, S, H); // h += ff @ W2
    }

    // ---- final LN, pooling, heads ----
    ln_kernel<<<MS, 256, 0, stream>>>(h, fin_ln_w, fin_ln_b, bigf, nullptr, D, 0, S);
    pool_last<<<(B * D + 255) / 256, 256, 0, stream>>>(bigf, pooled, B, S, D);

    float* out = (float*)d_out;                    // [cls 16*3 | reg 16 | conf 16]
    dense_small<<<(B * (D / 2) + 127) / 128, 128, 0, stream>>>(pooled, cls_w1, cls_b1, hid, B, D, D / 2, 1);
    dense_small<<<1, 64, 0, stream>>>(hid, cls_w2, cls_b2, out, B, D / 2, 3, 0);
    dense_small<<<(B * (D / 2) + 127) / 128, 128, 0, stream>>>(pooled, reg_w1, reg_b1, hid, B, D, D / 2, 1);
    dense_small<<<1, 32, 0, stream>>>(hid, reg_w2, reg_b2, out + B * 3, B, D / 2, 1, 0);
    dense_small<<<(B * (D / 4) + 127) / 128, 128, 0, stream>>>(pooled, conf_w1, conf_b1, hid, B, D, D / 4, 1);
    dense_small<<<1, 32, 0, stream>>>(hid, conf_w2, conf_b2, out + B * 4, B, D / 4, 1, 2);
}
